// LSTM_40278203302674
// MI455X (gfx1250) — compile-verified, matrix operand order checked by reading, not run on hardware
//
#include <hip/hip_runtime.h>

// ---------------- types ----------------
typedef __bf16 bf16_t;
typedef bf16_t v16bf __attribute__((ext_vector_type(16)));
typedef float  v8f   __attribute__((ext_vector_type(8)));
typedef unsigned v4u __attribute__((ext_vector_type(4)));
typedef int      v8i __attribute__((ext_vector_type(8)));
typedef int      v4i __attribute__((ext_vector_type(4)));
typedef unsigned short u16;

union FragBF {
    v16bf v;
    uint4 u[2];
};

__device__ __forceinline__ u16 f2bf(float f) {
    unsigned u = __float_as_uint(f);
    unsigned r = u + 0x7FFFu + ((u >> 16) & 1u);   // round-to-nearest-even
    return (u16)(r >> 16);
}
__device__ __forceinline__ float bf2f(u16 s) {
    return __uint_as_float(((unsigned)s) << 16);
}

#define LOG2E 1.4426950408889634f

// Problem constants
#define BATCH 128
#define TSTEPS 256
#define HDIM 1024
#define NCLS 10
#define NWG 32          // persistent workgroups; each owns 128 gate-rows (32 h-units x 4 gates)
#define NTHR 256        // 8 wave32 per WG

// Workspace layout (bytes):
//   [0)        barrier counter (u32), padded to 256
//   [256)      h double buffer: 2 * 128*1024 bf16 = 512 KB
//   [256+512K) swizzled bf16 weights: 32 WG * 256 frag * 32 lane * 16 bf16 = 8 MB
#define WS_HOFF   256
#define WS_WOFF   (256 + 2*BATCH*HDIM*2)
#define HBUF_ELEMS (BATCH*HDIM)          // per buffer
#define WG_WELEMS  (256*32*16)           // bf16 elems of weight per WG = 131072 (256KB)

// ---------------- prologue: init h buffers + barrier counter ----------------
__global__ __launch_bounds__(256) void lstm_init(u16* hbuf, unsigned* counter) {
    int gid = blockIdx.x * 256 + threadIdx.x;
    if (gid == 0) *counter = 0u;
    for (int i = gid; i < 2 * HBUF_ELEMS; i += 256 * 256) hbuf[i] = 0;
}

// ---------------- prologue: pack weights bf16, fragment-major swizzle ----------------
// Row interleave: global row r = 4*hunit + gate  (gate: 0=g,1=i,2=f,3=o).
// WG wg owns rows [128*wg, 128*wg+128) -> hunits [32*wg, 32*wg+32), all 4 gates.
// B-fragment (K32 x N16) per-lane layout mirrors the ISA 16-bit A layout:
//   lane<16:  K = kc*32 + {0..7, 16..23};  lane>=16: K = kc*32 + {8..15, 24..31}
//   n = nt*16 + (lane&15)
// Each lane's 16 bf16 stored contiguously (32B) so the main kernel does one v16bf ds load.
__global__ __launch_bounds__(256) void lstm_pack(
    const float* __restrict__ Wgh, const float* __restrict__ Wih,
    const float* __restrict__ Wfh, const float* __restrict__ Woh,
    u16* __restrict__ wsw)
{
    int id = blockIdx.x * 256 + threadIdx.x;   // 0 .. 262143  (32WG * 256frag * 32lane)
    int wg   = id >> 13;
    int rem  = id & 8191;
    int frag = rem >> 5;       // kc*8 + nt
    int lane = rem & 31;
    int kc = frag >> 3;
    int nt = frag & 7;
    int n_local = nt * 16 + (lane & 15);
    int row  = wg * 128 + n_local;
    int gate = row & 3;
    int hu   = row >> 2;
    const float* W = (gate == 0) ? Wgh : (gate == 1) ? Wih : (gate == 2) ? Wfh : Woh;
    const float* src = W + (size_t)hu * HDIM + kc * 32 + (lane >> 4) * 8;
    u16* dst = wsw + (size_t)id * 16;
#pragma unroll
    for (int e = 0; e < 8; ++e) dst[e]     = f2bf(src[e]);
#pragma unroll
    for (int e = 0; e < 8; ++e) dst[8 + e] = f2bf(src[16 + e]);
}

// ---------------- main persistent LSTM kernel ----------------
__global__ __launch_bounds__(NTHR, 1) void lstm_main(
    const float* __restrict__ x,
    const float* __restrict__ Wgx, const float* __restrict__ bg,
    const float* __restrict__ Wix, const float* __restrict__ bi,
    const float* __restrict__ Wfx, const float* __restrict__ bfv,
    const float* __restrict__ Wox, const float* __restrict__ bo,
    const float* __restrict__ Wph, const float* __restrict__ bp,
    const u16* __restrict__ wsw, u16* __restrict__ hbuf,
    unsigned* __restrict__ counter, float* __restrict__ out)
{
    extern __shared__ char smem[];
    u16*   wlds  = (u16*)smem;                       // 256 KB swizzled weights
    float* gates = (float*)(smem + WG_WELEMS * 2);   // 64 KB: gates[batch][128 rows]

    const int wg  = blockIdx.x;
    const int tid = threadIdx.x;

    // ---- stage this WG's 256KB weight slice into LDS via the Tensor Data Mover ----
    // 2D descriptor: 2-byte elements, tile 8192 x 16 (= 131072 bf16 = 256 KB),
    // contiguous (stride == dim0), LDS dest offset 0, no multicast/pad/iterate.
    if (tid < 32) {   // one wave issues the DMA (EXEC is ignored by TDM; branch is wave-uniform)
        unsigned long long ga = (unsigned long long)(wsw + (size_t)wg * WG_WELEMS);
        v4u g0;
        g0[0] = 1u;                                            // count=1, user mode
        g0[1] = 0u;                                            // lds_addr = 0 (dynamic LDS base)
        g0[2] = (unsigned)(ga & 0xFFFFFFFFull);                // global_addr[31:0]
        g0[3] = (unsigned)((ga >> 32) & 0x01FFFFFFull)         // global_addr[56:32]
              | (2u << 30);                                    // type = 2 ("image")
        v8i g1;
        g1[0] = (int)(1u << 16);          // workgroup_mask=0, data_size=1 (2 bytes)
        g1[1] = (int)(8192u << 16);       // tensor_dim0[15:0] in bits[63:48]
        g1[2] = (int)(16u << 16);         // tensor_dim0 hi=0, tensor_dim1[15:0]=16
        g1[3] = (int)(8192u << 16);       // tensor_dim1 hi=0, tile_dim0=8192
        g1[4] = (int)(16u);               // tile_dim1=16, tile_dim2=0
        g1[5] = (int)(8192u);             // tensor_dim0_stride[31:0]
        g1[6] = 0;                        // stride0 hi, tensor_dim1_stride lo
        g1[7] = 0;                        // tensor_dim1_stride hi
        v4i gz = {0, 0, 0, 0};
#if __clang_major__ >= 23
        v8i gx = {0, 0, 0, 0, 0, 0, 0, 0};
        __builtin_amdgcn_tensor_load_to_lds(g0, g1, gz, gz, gx, 0);
#else
        __builtin_amdgcn_tensor_load_to_lds(g0, g1, gz, gz, 0);
#endif
        __builtin_amdgcn_s_wait_tensorcnt(0);
    }

    const int wv    = tid >> 5;        // wave 0..7 -> batch tile [16*wv, 16*wv+16)
    const int lane  = tid & 31;
    const int l16   = lane & 15;
    const int lhalf = lane >> 4;
    const int gate  = lane & 3;        // per-lane constant under interleaved row order
    const int mbase = wv * 16 + lhalf * 8;   // batch base for this lane's 8 C rows

    // Activation constants (tanh(x) = 2*sigmoid(2x)-1):
    //   val = oa * rcp(1 + exp2(escale*pre)) + ob
    const float escale = (gate == 0) ? (-2.f * LOG2E) : (-LOG2E);
    const float oa     = (gate == 0) ? 2.f  : 1.f;
    const float ob     = (gate == 0) ? -1.f : 0.f;

    // per-tile row constants (hoisted out of the time loop)
    float wxv[8], bv[8];
#pragma unroll
    for (int nt = 0; nt < 8; ++nt) {
        int row = wg * 128 + nt * 16 + l16;
        int hu  = row >> 2;
        const float* WxSel = (gate == 0) ? Wgx : (gate == 1) ? Wix : (gate == 2) ? Wfx : Wox;
        const float* bSel  = (gate == 0) ? bg  : (gate == 1) ? bi  : (gate == 2) ? bfv : bo;
        wxv[nt] = WxSel[hu];
        bv[nt]  = bSel[hu];
    }

    // c-state registers for the update phase:
    //   thread -> (batch = tid>>1, hidden local [ (tid&1)*16, +16) )
    float cst[16];
#pragma unroll
    for (int j = 0; j < 16; ++j) cst[j] = 0.f;
    const int b2  = tid >> 1;
    const int lh2 = (tid & 1) * 16;

    __syncthreads();   // LDS weights ready (TDM completed by wave 0)

    const u16* bbase = wlds + lane * 16;   // per-lane base into fragment-major weights

    for (int t = 0; t < TSTEPS; ++t) {
        const u16* hread  = hbuf + (size_t)(t & 1) * HBUF_ELEMS;
        u16*       hwrite = hbuf + (size_t)((t + 1) & 1) * HBUF_ELEMS;

        // ---- GEMM: pre[batch 16][rows 128] accumulated in 8 WMMA tiles ----
        v8f acc[8];
#pragma unroll
        for (int nt = 0; nt < 8; ++nt)
#pragma unroll
            for (int e = 0; e < 8; ++e) acc[nt][e] = 0.f;

        const u16* arow = hread + (size_t)(wv * 16 + l16) * HDIM + lhalf * 8;

        // software pipeline: K unrolled by 2 with ping-pong A registers (no copies),
        // all 8 B fragments of a half-chunk preloaded into distinct registers so the
        // ds_loads clause together and WMMAs only take partial dscnt waits.
        FragBF a0, a1;
        a0.u[0] = *(const uint4*)(arow);
        a0.u[1] = *(const uint4*)(arow + 16);
#pragma unroll 1
        for (int kc = 0; kc < 32; kc += 2) {
            // issue A(kc+1) before the a0 chain; its wait lands after 8 WMMAs
            a1.u[0] = *(const uint4*)(arow + (kc + 1) * 32);
            a1.u[1] = *(const uint4*)(arow + (kc + 1) * 32 + 16);
            __builtin_prefetch(arow + (kc + 2) * 32, 0, 1);
            {
                FragBF b[8];
#pragma unroll
                for (int nt = 0; nt < 8; ++nt)
                    b[nt].v = *(const v16bf*)(bbase + (kc * 8 + nt) * 512);
#pragma unroll
                for (int nt = 0; nt < 8; ++nt)
                    acc[nt] = __builtin_amdgcn_wmma_f32_16x16x32_bf16(
                        false, a0.v, false, b[nt].v, (short)0, acc[nt], false, false);
            }
            // issue A(kc+2) before the a1 chain (clamped on last iteration; address stays valid)
            const int knxt = (kc + 2 < 32) ? (kc + 2) : 0;
            a0.u[0] = *(const uint4*)(arow + knxt * 32);
            a0.u[1] = *(const uint4*)(arow + knxt * 32 + 16);
            {
                FragBF b[8];
#pragma unroll
                for (int nt = 0; nt < 8; ++nt)
                    b[nt].v = *(const v16bf*)(bbase + ((kc + 1) * 8 + nt) * 512);
#pragma unroll
                for (int nt = 0; nt < 8; ++nt)
                    acc[nt] = __builtin_amdgcn_wmma_f32_16x16x32_bf16(
                        false, a1.v, false, b[nt].v, (short)0, acc[nt], false, false);
            }
        }

        // ---- x contribution + bias + activation, exchange via LDS ----
        float xv[8];
#pragma unroll
        for (int r = 0; r < 8; ++r) xv[r] = x[(size_t)(mbase + r) * TSTEPS + t];
#pragma unroll
        for (int nt = 0; nt < 8; ++nt) {
            const int ncol = nt * 16 + l16;
#pragma unroll
            for (int r = 0; r < 8; ++r) {
                float pre = acc[nt][r] + xv[r] * wxv[nt] + bv[nt];
                float e   = __builtin_amdgcn_exp2f(escale * pre);
                float val = oa * __builtin_amdgcn_rcpf(1.f + e) + ob;
                gates[(mbase + r) * 128 + ncol] = val;
            }
        }
        __syncthreads();

        // ---- state update: c = g*i + c*f ; h = tanh(c)*o ----
#pragma unroll 4
        for (int j = 0; j < 16; ++j) {
            const float4 gi = *(const float4*)(gates + b2 * 128 + (lh2 + j) * 4);
            float cn = gi.x * gi.y + cst[j] * gi.z;
            cst[j] = cn;
            float e  = __builtin_amdgcn_exp2f(-2.f * LOG2E * cn);
            float th = 2.f * __builtin_amdgcn_rcpf(1.f + e) - 1.f;
            float hv = th * gi.w;
            hwrite[(size_t)b2 * HDIM + wg * 32 + lh2 + j] = f2bf(hv);
        }

        // ---- device-wide barrier (arrive + spin on monotonic counter) ----
        __threadfence();
        __syncthreads();   // also protects gates LDS reuse next iteration
        if (tid == 0) {
            __hip_atomic_fetch_add(counter, 1u, __ATOMIC_RELEASE, __HIP_MEMORY_SCOPE_AGENT);
            const unsigned target = (unsigned)NWG * (unsigned)(t + 1);
            while (__hip_atomic_load(counter, __ATOMIC_ACQUIRE, __HIP_MEMORY_SCOPE_AGENT) < target)
                __builtin_amdgcn_s_sleep(2);
        }
        __syncthreads();
    }

    // ---- epilogue: out = h_final @ W_ph + bias_p  (final h is in buffer 0) ----
    const u16* hfin = hbuf;   // TSTEPS even -> last write went to buffer 0
    const int b   = wg * 4 + (tid >> 6);
    const int cls = tid & 63;
    if (cls < NCLS) {
        float s = 0.f;
        const u16* hr = hfin + (size_t)b * HDIM;
        for (int k = 0; k < HDIM; ++k)
            s += bf2f(hr[k]) * Wph[(size_t)k * NCLS + cls];
        out[b * NCLS + cls] = s + bp[cls];
    }
}

// ---------------- host launch ----------------
extern "C" void kernel_launch(void* const* d_in, const int* in_sizes, int n_in,
                              void* d_out, int out_size, void* d_ws, size_t ws_size,
                              hipStream_t stream) {
    const float* x    = (const float*)d_in[0];
    const float* Wgx  = (const float*)d_in[1];
    const float* Wgh  = (const float*)d_in[2];
    const float* bg   = (const float*)d_in[3];
    const float* Wix  = (const float*)d_in[4];
    const float* Wih  = (const float*)d_in[5];
    const float* bi   = (const float*)d_in[6];
    const float* Wfx  = (const float*)d_in[7];
    const float* Wfh  = (const float*)d_in[8];
    const float* bfv  = (const float*)d_in[9];
    const float* Wox  = (const float*)d_in[10];
    const float* Woh  = (const float*)d_in[11];
    const float* bo   = (const float*)d_in[12];
    const float* Wph  = (const float*)d_in[13];
    const float* bp   = (const float*)d_in[14];

    char* ws = (char*)d_ws;
    unsigned* counter = (unsigned*)ws;
    u16* hbuf = (u16*)(ws + WS_HOFF);
    u16* wsw  = (u16*)(ws + WS_WOFF);

    lstm_init<<<256, 256, 0, stream>>>(hbuf, counter);
    lstm_pack<<<1024, 256, 0, stream>>>(Wgh, Wih, Wfh, Woh, wsw);

    const size_t shmem = (size_t)WG_WELEMS * 2 + 64 * 1024;   // 256KB weights + 64KB gates = 320KB
    lstm_main<<<NWG, NTHR, shmem, stream>>>(
        x, Wgx, bg, Wix, bi, Wfx, bfv, Wox, bo, Wph, bp,
        wsw, hbuf, counter, (float*)d_out);
}
